// AdaptiveCausalAttention_31756988187209
// MI455X (gfx1250) — matrix operand order repaired and checked by reading, not compiled, not run on hardware
//
#include <hip/hip_runtime.h>
#include <hip/hip_bf16.h>
#include <math.h>

typedef __bf16 bf16;
typedef __attribute__((ext_vector_type(16))) __bf16 v16bf;
typedef __attribute__((ext_vector_type(8)))  __bf16 v8bf;
typedef __attribute__((ext_vector_type(8)))  float  v8f;

#define T_DIM 1024
#define C_DIM 1024
#define B_DIM 4
#define H_DIM 16
#define D_HEAD 64
#define NEG_BIG (-1.0e30f)
#define TWO_PI 6.2831853071795864f

static __device__ __forceinline__ int imin(int a, int b) { return a < b ? a : b; }

static __device__ __forceinline__ float sigmoidf_dev(float x) {
    return 1.0f / (1.0f + __expf(-x));
}

static __device__ __forceinline__ v8f wmma_bf16(v16bf a, v16bf b, v8f c) {
    // D = A(16x32 bf16) * B(32x16 bf16) + C(16x16 f32)
    return __builtin_amdgcn_wmma_f32_16x16x32_bf16(
        /*neg_a=*/false, a, /*neg_b=*/false, b,
        /*c_mod=*/(short)0, c, /*reuse_a=*/false, /*reuse_b=*/false);
}

// Build a 16-element bf16 fragment from two contiguous 8-element (16B) runs.
static __device__ __forceinline__ v16bf make_frag(const bf16* p_lo, const bf16* p_hi) {
    v8bf lo = *(const v8bf*)p_lo;
    v8bf hi = *(const v8bf*)p_hi;
    v16bf a;
#pragma unroll
    for (int i = 0; i < 8; ++i) { a[i] = lo[i]; a[i + 8] = hi[i]; }
    return a;
}

// A-matrix fragment (16x32 MxK, bf16): lane half h holds runs k=8h..8h+7 and 16+8h..16+8h+7.
static __device__ __forceinline__ v16bf fragA(const bf16* row, int k0, int h) {
    return make_frag(row + k0 + 8 * h, row + k0 + 16 + 8 * h);
}
// B-matrix fragment (32x16 KxN, bf16) read from N-major storage: lane half h holds runs k=8h..8h+7 and 16+8h..16+8h+7.
static __device__ __forceinline__ v16bf fragB(const bf16* row, int k0, int h) {
    return make_frag(row + k0 + 8 * h, row + k0 + 16 + 8 * h);
}

static __device__ __forceinline__ v8f vzero8f() {
    v8f z;
#pragma unroll
    for (int i = 0; i < 8; ++i) z[i] = 0.0f;
    return z;
}

// ---------------------------------------------------------------------------
// Elementwise f32 -> bf16 convert
__global__ void k_cvt_bf16(const float* __restrict__ in, bf16* __restrict__ out, int n) {
    int i = blockIdx.x * blockDim.x + threadIdx.x;
    if (i < n) out[i] = (bf16)in[i];
}

// Transpose+convert: in [rowsK x colsN] f32 row-major -> out [colsN x rowsK] bf16 row-major
__global__ void k_transpose_bf16(const float* __restrict__ in, bf16* __restrict__ out,
                                 int rowsK, int colsN) {
    int i = blockIdx.x * blockDim.x + threadIdx.x;
    if (i < rowsK * colsN) {
        int n = i / rowsK;
        int k = i - n * rowsK;
        out[i] = (bf16)in[(size_t)k * colsN + n];
    }
}

// ---------------------------------------------------------------------------
// GEMM1: qkv = x @ Wqkv.  A = xb [4096 x 1024] bf16, Bt = WqkvT [3072 x 1024] bf16.
// Epilogue scatters to q[B,H,T,D], k[B,H,T,D], vT[B,H,D,T] as bf16.
__global__ __launch_bounds__(256) void k_gemm_qkv(const bf16* __restrict__ A,
                                                  const bf16* __restrict__ Bt,
                                                  bf16* __restrict__ qb,
                                                  bf16* __restrict__ kb,
                                                  bf16* __restrict__ vT) {
    const int K = C_DIM;
    int lane = threadIdx.x & 31;
    int wid  = threadIdx.x >> 5;
    int h = lane >> 4, l = lane & 15;
    int wm = wid >> 2, wn = wid & 3;
    int rowbase = blockIdx.x * 64 + wm * 32;
    int colbase = blockIdx.y * 128 + wn * 32;

    const bf16* arow0 = A + (size_t)(rowbase + l) * K;
    const bf16* arow1 = A + (size_t)(rowbase + 16 + l) * K;
    const bf16* brow0 = Bt + (size_t)(colbase + l) * K;
    const bf16* brow1 = Bt + (size_t)(colbase + 16 + l) * K;

    v8f acc00 = vzero8f(), acc01 = vzero8f(), acc10 = vzero8f(), acc11 = vzero8f();

    for (int kk = 0; kk < K; kk += 32) {
        __builtin_prefetch(arow0 + kk + 256, 0, 0);
        __builtin_prefetch(brow0 + kk + 256, 0, 0);
        v16bf a0 = fragA(arow0, kk, h);
        v16bf a1 = fragA(arow1, kk, h);
        v16bf b0 = fragB(brow0, kk, h);
        v16bf b1 = fragB(brow1, kk, h);
        acc00 = wmma_bf16(a0, b0, acc00);
        acc01 = wmma_bf16(a0, b1, acc01);
        acc10 = wmma_bf16(a1, b0, acc10);
        acc11 = wmma_bf16(a1, b1, acc11);
    }

    const v8f* accs[4] = { &acc00, &acc01, &acc10, &acc11 };
#pragma unroll
    for (int i = 0; i < 2; ++i)
#pragma unroll
        for (int j = 0; j < 2; ++j) {
            const v8f& acc = *accs[i * 2 + j];
#pragma unroll
            for (int r = 0; r < 8; ++r) {
                int m = rowbase + i * 16 + r + 8 * h;   // 0..4095 (b*T + t)
                int n = colbase + j * 16 + l;           // 0..3071
                int b = m >> 10, t = m & 1023;
                int sel = n >> 10, nn = n & 1023;
                int hh = nn >> 6, d = nn & 63;
                bf16 bv = (bf16)acc[r];
                if (sel == 0)
                    qb[(((size_t)(b * H_DIM + hh) * T_DIM + t) * D_HEAD) + d] = bv;
                else if (sel == 1)
                    kb[(((size_t)(b * H_DIM + hh) * T_DIM + t) * D_HEAD) + d] = bv;
                else
                    vT[((size_t)(b * H_DIM + hh) * D_HEAD + d) * T_DIM + t] = bv;
            }
        }
}

// ---------------------------------------------------------------------------
// Fused flash attention with analytic span + triangle-wave log-mask.
// One wave per 16-row q-tile; 4 waves (128 threads) per block.
__global__ __launch_bounds__(128) void k_attn(const bf16* __restrict__ qb,
                                              const bf16* __restrict__ kb,
                                              const bf16* __restrict__ vT,
                                              const float* __restrict__ span_params,
                                              const float* __restrict__ period_w,
                                              const float* __restrict__ ratio_w,
                                              bf16* __restrict__ yb) {
    __shared__ bf16 plds[4][16 * 32];   // per-wave P staging (C-layout -> A-layout)

    int lane = threadIdx.x & 31;
    int wid  = threadIdx.x >> 5;
    int h16 = lane >> 4, l = lane & 15;

    int gt = blockIdx.x * 4 + wid;              // global q-tile id (0..4095)
    int bh = gt >> 6;                            // (b*H + h), 64 q-tiles per head
    int qt = gt & 63;
    int hh = bh & 15;
    int b  = bh >> 4;
    int t0 = qt * 16;

    // Per-head mask parameters
    float span   = 1024.0f * sigmoidf_dev(span_params[hh]);
    float period = 2.0f + 62.0f * sigmoidf_dev(period_w[hh]);
    float ratio  = -0.25f + 0.5f * sigmoidf_dev(ratio_w[hh]);
    float amp    = 0.25f * period;
    float off    = period * ratio;
    float inv_period_2pi = TWO_PI / period;
    const float C1 = 16.0f / (float)(M_PI * M_PI);
    const float C3 = C1 / 9.0f;
    const float C5 = C1 / 25.0f;

    // Q A-fragments (16x32 per 32-wide d slice), resident for whole key loop
    const bf16* qrow = qb + ((size_t)bh * T_DIM + t0 + l) * D_HEAD;
    v16bf aq0 = fragA(qrow, 0, h16);
    v16bf aq1 = fragA(qrow, 32, h16);

    v8f O0 = vzero8f(), O1 = vzero8f(), O2 = vzero8f(), O3 = vzero8f();
    float Mr[8], Lr[8];
#pragma unroll
    for (int r = 0; r < 8; ++r) { Mr[r] = NEG_BIG; Lr[r] = 0.0f; }

    const bf16* kbase = kb + (size_t)bh * T_DIM * D_HEAD;
    const bf16* vbase = vT + (size_t)bh * D_HEAD * T_DIM;
    bf16* myp = plds[wid];
    int tmax = t0 + 15;

    for (int k0 = 0; k0 <= tmax; k0 += 32) {
        // ---- S = Q K^T for two 16-key subtiles ----
        v8f S[2];
#pragma unroll
        for (int ns = 0; ns < 2; ++ns) {
            int key  = k0 + ns * 16 + l;
            int keyc = imin(key, T_DIM - 1);
            const bf16* krow = kbase + (size_t)keyc * D_HEAD;
            v16bf bk0 = fragB(krow, 0, h16);
            v16bf bk1 = fragB(krow, 32, h16);
            v8f s = vzero8f();
            s = wmma_bf16(aq0, bk0, s);
            s = wmma_bf16(aq1, bk1, s);
            S[ns] = s;
        }

        // ---- masks + online softmax over the 32-key block ----
        float w[2][8];
#pragma unroll
        for (int ns = 0; ns < 2; ++ns) {
            float kpos = (float)(k0 + ns * 16 + l);
#pragma unroll
            for (int r = 0; r < 8; ++r) {
                float qpos = (float)(t0 + r + 8 * h16);
                float rel = qpos - kpos;
                float wv = NEG_BIG;
                if (rel >= 0.0f) {
                    float mspan = fminf(fmaxf((32.0f - rel + span) * (1.0f / 32.0f), 0.0f), 1.0f);
                    float ph = rel * inv_period_2pi;
                    ph = ph - floorf(ph * (1.0f / TWO_PI)) * TWO_PI;
                    float wave = C1 * __cosf(ph) + C3 * __cosf(3.0f * ph) + C5 * __cosf(5.0f * ph);
                    wave = wave * amp * 0.5f + 0.5f + off;
                    float mtri = fminf(fmaxf(wave, 0.0f), 1.0f);
                    if (fminf(mspan, mtri) > 1e-6f)
                        wv = S[ns][r] * 0.125f
                           + __logf(fmaxf(mspan, 1e-6f))
                           + __logf(fmaxf(mtri, 1e-6f));
                }
                w[ns][r] = wv;
            }
        }

#pragma unroll
        for (int r = 0; r < 8; ++r) {
            float rm = fmaxf(w[0][r], w[1][r]);
#pragma unroll
            for (int mk = 1; mk < 16; mk <<= 1)
                rm = fmaxf(rm, __shfl_xor(rm, mk, 32));
            float Mnew = fmaxf(Mr[r], rm);
            float alpha, p0, p1;
            if (Mnew < -0.5e30f) {                 // fully-masked so far: keep zeros
                alpha = 1.0f; p0 = 0.0f; p1 = 0.0f;
            } else {
                alpha = __expf(Mr[r] - Mnew);
                p0 = __expf(w[0][r] - Mnew);
                p1 = __expf(w[1][r] - Mnew);
            }
            Mr[r] = Mnew;
            float rs = p0 + p1;
#pragma unroll
            for (int mk = 1; mk < 16; mk <<= 1)
                rs += __shfl_xor(rs, mk, 32);
            Lr[r] = Lr[r] * alpha + rs;
            O0[r] *= alpha; O1[r] *= alpha; O2[r] *= alpha; O3[r] *= alpha;
            int m = r + 8 * h16;                   // tile row (C-layout)
            myp[m * 32 + l]      = (bf16)p0;
            myp[m * 32 + 16 + l] = (bf16)p1;
        }

        asm volatile("s_wait_dscnt 0" ::: "memory");

        // ---- O += P V ----
        v16bf ap = fragA(myp + l * 32, 0, h16);    // re-read P in A-layout
        int s0 = imin(k0 + 16 * h16, T_DIM - 16);  // clamp tail (p==0 there)
        {
            v16bf bv = fragB(vbase + (size_t)(0 * 16 + l) * T_DIM - 16 * h16, s0, h16);
            O0 = wmma_bf16(ap, bv, O0);
        }
        {
            v16bf bv = fragB(vbase + (size_t)(1 * 16 + l) * T_DIM - 16 * h16, s0, h16);
            O1 = wmma_bf16(ap, bv, O1);
        }
        {
            v16bf bv = fragB(vbase + (size_t)(2 * 16 + l) * T_DIM - 16 * h16, s0, h16);
            O2 = wmma_bf16(ap, bv, O2);
        }
        {
            v16bf bv = fragB(vbase + (size_t)(3 * 16 + l) * T_DIM - 16 * h16, s0, h16);
            O3 = wmma_bf16(ap, bv, O3);
        }
    }

    // ---- epilogue: y[b, t, hh*64 + d] = O / L  (bf16) ----
    const v8f* Os[4] = { &O0, &O1, &O2, &O3 };
#pragma unroll
    for (int j = 0; j < 4; ++j) {
        const v8f& O = *Os[j];
#pragma unroll
        for (int r = 0; r < 8; ++r) {
            int qpos = t0 + r + 8 * h16;
            float denom = Lr[r];
            float val = (denom > 0.0f) ? O[r] / denom : 0.0f;
            yb[((size_t)(b * T_DIM + qpos) * H_DIM + hh) * D_HEAD + j * 16 + l] = (bf16)val;
        }
    }
}

// ---------------------------------------------------------------------------
// GEMM3: out = y @ Wproj.  A = yb [4096 x 1024] bf16, Bt = WprojT [1024 x 1024] bf16.
__global__ __launch_bounds__(256) void k_gemm_proj(const bf16* __restrict__ A,
                                                   const bf16* __restrict__ Bt,
                                                   float* __restrict__ out) {
    const int K = C_DIM;
    int lane = threadIdx.x & 31;
    int wid  = threadIdx.x >> 5;
    int h = lane >> 4, l = lane & 15;
    int wm = wid >> 2, wn = wid & 3;
    int rowbase = blockIdx.x * 64 + wm * 32;
    int colbase = blockIdx.y * 128 + wn * 32;

    const bf16* arow0 = A + (size_t)(rowbase + l) * K;
    const bf16* arow1 = A + (size_t)(rowbase + 16 + l) * K;
    const bf16* brow0 = Bt + (size_t)(colbase + l) * K;
    const bf16* brow1 = Bt + (size_t)(colbase + 16 + l) * K;

    v8f acc00 = vzero8f(), acc01 = vzero8f(), acc10 = vzero8f(), acc11 = vzero8f();

    for (int kk = 0; kk < K; kk += 32) {
        __builtin_prefetch(arow0 + kk + 256, 0, 0);
        __builtin_prefetch(brow0 + kk + 256, 0, 0);
        v16bf a0 = fragA(arow0, kk, h);
        v16bf a1 = fragA(arow1, kk, h);
        v16bf b0 = fragB(brow0, kk, h);
        v16bf b1 = fragB(brow1, kk, h);
        acc00 = wmma_bf16(a0, b0, acc00);
        acc01 = wmma_bf16(a0, b1, acc01);
        acc10 = wmma_bf16(a1, b0, acc10);
        acc11 = wmma_bf16(a1, b1, acc11);
    }

    const v8f* accs[4] = { &acc00, &acc01, &acc10, &acc11 };
#pragma unroll
    for (int i = 0; i < 2; ++i)
#pragma unroll
        for (int j = 0; j < 2; ++j) {
            const v8f& acc = *accs[i * 2 + j];
#pragma unroll
            for (int r = 0; r < 8; ++r) {
                int m = rowbase + i * 16 + r + 8 * h;
                int n = colbase + j * 16 + l;
                out[(size_t)m * C_DIM + n] = acc[r];
            }
        }
}

// ---------------------------------------------------------------------------
// span_loss (scalar): one wave.
__global__ void k_span_loss(const float* __restrict__ sp, const float* __restrict__ pw,
                            const float* __restrict__ rw, float* __restrict__ out) {
    int l = threadIdx.x;
    float v = 0.0f;
    if (l < 16) {
        float span   = 1024.0f * sigmoidf_dev(sp[l]);
        float period = 2.0f + 62.0f * sigmoidf_dev(pw[l]);
        float ratio  = -0.25f + 0.5f * sigmoidf_dev(rw[l]);
        float amp    = 0.25f * period;
        float off    = period * ratio;
        float base   = 1.0f / period + 2.0f * ratio + 0.5f;
        float lt     = (base < 1.0f) ? base : 1.0f + (0.5f + off - amp);
        v = (span + 32.0f) * lt;
    }
#pragma unroll
    for (int mk = 1; mk < 32; mk <<= 1) v += __shfl_xor(v, mk, 32);
    if (l == 0) out[0] = 1e-4f * v / 16.0f;
}

// ---------------------------------------------------------------------------
extern "C" void kernel_launch(void* const* d_in, const int* in_sizes, int n_in,
                              void* d_out, int out_size, void* d_ws, size_t ws_size,
                              hipStream_t stream) {
    (void)in_sizes; (void)n_in; (void)out_size; (void)ws_size;
    const float* x     = (const float*)d_in[0];
    const float* Wqkv  = (const float*)d_in[1];
    const float* Wproj = (const float*)d_in[2];
    const float* sp    = (const float*)d_in[3];
    const float* pw    = (const float*)d_in[4];
    const float* rw    = (const float*)d_in[5];
    float* out = (float*)d_out;

    char* ws = (char*)d_ws;
    bf16* xb     = (bf16*)(ws);                      //  8 MB: x as bf16
    bf16* WqkvT  = (bf16*)(ws + ( 8ull << 20));      //  6 MB: Wqkv^T bf16
    bf16* WprojT = (bf16*)(ws + (14ull << 20));      //  2 MB: Wproj^T bf16
    bf16* qb     = (bf16*)(ws + (16ull << 20));      //  8 MB: q [B,H,T,D]
    bf16* kb     = (bf16*)(ws + (24ull << 20));      //  8 MB: k [B,H,T,D]
    bf16* vT     = (bf16*)(ws + (32ull << 20));      //  8 MB: v^T [B,H,D,T]
    bf16* yb     = (bf16*)(ws + (40ull << 20));      //  8 MB: y [B,T,C]

    const int nx = B_DIM * T_DIM * C_DIM;            // 4M
    k_cvt_bf16<<<nx / 256, 256, 0, stream>>>(x, xb, nx);
    k_transpose_bf16<<<(3 * 1024 * 1024) / 256, 256, 0, stream>>>(Wqkv, WqkvT, 1024, 3072);
    k_transpose_bf16<<<(1024 * 1024) / 256, 256, 0, stream>>>(Wproj, WprojT, 1024, 1024);

    dim3 g1(4096 / 64, 3072 / 128);
    k_gemm_qkv<<<g1, 256, 0, stream>>>(xb, WqkvT, qb, kb, vT);

    k_attn<<<(B_DIM * H_DIM * (T_DIM / 16)) / 4, 128, 0, stream>>>(
        qb, kb, vT, sp, pw, rw, yb);

    dim3 g3(4096 / 64, 1024 / 128);
    k_gemm_proj<<<g3, 256, 0, stream>>>(yb, WprojT, out);

    k_span_loss<<<1, 32, 0, stream>>>(sp, pw, rw, out + (size_t)nx);
}
